// LayoutLmv2SelfAttention_6296422056369
// MI455X (gfx1250) — matrix operand order repaired and checked by reading, not compiled, not run on hardware
//
#include <hip/hip_runtime.h>

// LayoutLMv2 self-attention for MI455X (gfx1250, wave32).
// bf16 WMMA (v_wmma_f32_16x16x32_bf16) for QKV GEMM, QK^T and PV; fp32
// accumulation; flash-style online softmax. CDNA5 async global->LDS staging
// (ASYNCcnt) with LDS double-buffering in the GEMM.
// Workspace (assumes ws_size >= ~73.5 MB):
//   hb bf16 16MB | wb bf16 6MB | qh/kh/vh bf16 16MB each.
//
// Probe results (round 2): global_load_async_to_lds_b128 / s_wait_asynccnt /
// tensor_load_to_lds / cluster_load_async_to_lds_b128 AVAILABLE;
// ds/global transpose-load builtins NOT available on this toolchain.

#define B_  8
#define N_  1024
#define H_  16
#define D_  64
#define HD_ 1024

typedef __bf16 v16bf __attribute__((ext_vector_type(16)));
typedef float  v8f   __attribute__((ext_vector_type(8)));
typedef int    v4i   __attribute__((ext_vector_type(4)));

union Frag16 { v16bf v; uint4 q[2]; };

#if __has_builtin(__builtin_amdgcn_global_load_async_to_lds_b128) && \
    __has_builtin(__builtin_amdgcn_s_wait_asynccnt)
#define HAS_ASYNC 1
#else
#define HAS_ASYNC 0
#endif

#if HAS_ASYNC
// Signature (confirmed by clang diagnostic): (int4 AS1* gsrc, int4 AS3* ldst,
// imm offset, imm cpol)
static __device__ __forceinline__ void async_cp_b128(const void* g, void* l) {
  __builtin_amdgcn_global_load_async_to_lds_b128(
      (__attribute__((address_space(1))) v4i*)(g),
      (__attribute__((address_space(3))) v4i*)(l), 0, 0);
}
static __device__ __forceinline__ void wait_async_all() {
  __builtin_amdgcn_s_wait_asynccnt(0);
}
#else
static __device__ __forceinline__ void wait_async_all() {}
#endif

static __device__ __forceinline__ v8f wmma_bf16(v16bf a, v16bf b, v8f c) {
  // (neg_a, A, neg_b, B, c_mod, C, reuse_a, reuse_b)
  return __builtin_amdgcn_wmma_f32_16x16x32_bf16(false, a, false, b, (short)0, c,
                                                 false, false);
}

// ---------------------------------------------------------------------------
// fp32 -> bf16 conversion (8 elems/thread, 16B packed stores)
// ---------------------------------------------------------------------------
__global__ __launch_bounds__(256) void cvt_f32_bf16_kernel(
    const float* __restrict__ src, unsigned short* __restrict__ dst, int n) {
  int i = (blockIdx.x * 256 + threadIdx.x) * 8;
  if (i >= n) return;
  float4 f0 = *(const float4*)(src + i);
  float4 f1 = *(const float4*)(src + i + 4);
  union { __bf16 e[8]; uint4 u; } t;
  t.e[0] = (__bf16)f0.x; t.e[1] = (__bf16)f0.y;
  t.e[2] = (__bf16)f0.z; t.e[3] = (__bf16)f0.w;
  t.e[4] = (__bf16)f1.x; t.e[5] = (__bf16)f1.y;
  t.e[6] = (__bf16)f1.z; t.e[7] = (__bf16)f1.w;
  *(uint4*)(dst + i) = t.u;
}

// ---------------------------------------------------------------------------
// QKV projection: out[n,o] = sum_h hs[n,h]*w[o,h]; M=8192, N=3072, K=1024.
// 128(M) x 64(N) tile per workgroup, double-buffered async LDS staging.
// Epilogue fuses q/v bias + 1/sqrt(D) and scatters to head-major bf16 q/k/v.
// ---------------------------------------------------------------------------
__global__ __launch_bounds__(256) void qkv_gemm_kernel(
    const unsigned short* __restrict__ hb_u, const unsigned short* __restrict__ wb_u,
    const float* __restrict__ q_bias, const float* __restrict__ v_bias,
    unsigned short* __restrict__ qh_u, unsigned short* __restrict__ kh_u,
    unsigned short* __restrict__ vh_u) {
  const __bf16* hb = (const __bf16*)hb_u;
  const __bf16* wb = (const __bf16*)wb_u;

  __shared__ __attribute__((aligned(16))) __bf16 As[2][128 * 32];  // [m][k] 2x8KB
  __shared__ __attribute__((aligned(16))) __bf16 Bs2[2][64 * 32];  // [o][k] 2x4KB

  const int tid = threadIdx.x;
  const int wv = tid >> 5, lane = tid & 31, l16 = lane & 15, hf = lane >> 4;
  const int row0 = blockIdx.x * 128;
  const int col0 = blockIdx.y * 64;

  // stage one 128x32 A tile + 64x32 B tile into buffer `buf` (16B chunks)
  auto stage = [&](int buf, int k0) {
#if HAS_ASYNC
#pragma unroll
    for (int i = 0; i < 2; ++i) {
      const int c = tid + i * 256, r = c >> 2, cg = (c & 3) * 8;
      async_cp_b128(hb + (size_t)(row0 + r) * HD_ + k0 + cg, &As[buf][r * 32 + cg]);
    }
    {
      const int r = tid >> 2, cg = (tid & 3) * 8;
      async_cp_b128(wb + (size_t)(col0 + r) * HD_ + k0 + cg, &Bs2[buf][r * 32 + cg]);
    }
#else
    const int r0 = tid >> 2, cg0 = (tid & 3) * 8;
    const int c1 = tid + 256, r1 = c1 >> 2, cg1 = (c1 & 3) * 8;
    uint4 ta0 = *(const uint4*)(hb + (size_t)(row0 + r0) * HD_ + k0 + cg0);
    uint4 ta1 = *(const uint4*)(hb + (size_t)(row0 + r1) * HD_ + k0 + cg1);
    uint4 tb0 = *(const uint4*)(wb + (size_t)(col0 + r0) * HD_ + k0 + cg0);
    *(uint4*)(&As[buf][r0 * 32 + cg0]) = ta0;
    *(uint4*)(&As[buf][r1 * 32 + cg1]) = ta1;
    *(uint4*)(&Bs2[buf][r0 * 32 + cg0]) = tb0;
#endif
  };

  v8f acc[4] = {};

  stage(0, 0);
  wait_async_all();
  __syncthreads();

  int buf = 0;
  for (int k0 = 0; k0 < HD_; k0 += 32, buf ^= 1) {
    if (k0 + 32 < HD_) stage(buf ^ 1, k0 + 32);  // prefetch next tile

    Frag16 a;  // A-frag: lane<16 -> M=l16, K {0..7,16..23}; lane>=16 -> K+8
    const __bf16* ap = &As[buf][(wv * 16 + l16) * 32 + hf * 8];
    a.q[0] = *(const uint4*)(ap);
    a.q[1] = *(const uint4*)(ap + 16);

#pragma unroll
    for (int t = 0; t < 4; ++t) {
      Frag16 b;  // B-frag: N=l16 (+16t), K {0..7,16..23} (+8 for hf)
      const __bf16* bp = &Bs2[buf][(t * 16 + l16) * 32 + hf * 8];
      b.q[0] = *(const uint4*)(bp);
      b.q[1] = *(const uint4*)(bp + 16);
      acc[t] = wmma_bf16(a.v, b.v, acc[t]);
    }

    wait_async_all();
    __syncthreads();
  }

  const int seg = col0 >> 10;  // 0=q, 1=k, 2=v (64-col tile never straddles)
  __bf16* dst = (__bf16*)(seg == 0 ? qh_u : (seg == 1 ? kh_u : vh_u));
#pragma unroll
  for (int t = 0; t < 4; ++t) {
    const int o_loc = (col0 & 1023) + t * 16 + l16;  // offset within segment
    const int oh = o_loc >> 6, od = o_loc & 63;
    float bias = 0.f;
    if (seg == 0) bias = q_bias[o_loc];
    else if (seg == 2) bias = v_bias[o_loc];
#pragma unroll
    for (int r = 0; r < 8; ++r) {
      float x = acc[t][r];                     // D elem (M=r+8*hf, N=t*16+l16)
      if (seg == 0) x = (x + bias) * 0.125f;   // q: +bias then /sqrt(64)
      else if (seg == 2) x = x + bias;         // v: +bias
      const int ng = row0 + wv * 16 + r + hf * 8;
      const int nb = ng >> 10, nn = ng & 1023;
      dst[(((size_t)(nb * 16 + oh)) * N_ + nn) * D_ + od] = (__bf16)x;
    }
  }
}

// ---------------------------------------------------------------------------
// Fused flash attention: block = (q-tile of 128 rows) x (b,h). 8 waves,
// wave owns 16 q-rows. Per 32-key step: async-stage K[key][d], scatter-stage
// V^T[d][key], QK^T via 4 WMMAs, rel biases + mask (=1e-8 fill), online
// softmax (shfl_xor half-wave reductions), P through wave-private LDS to
// A-frag layout, PV via 4 WMMAs.
// ---------------------------------------------------------------------------
__global__ __launch_bounds__(256) void attn_kernel(
    const unsigned short* __restrict__ qh_u, const unsigned short* __restrict__ kh_u,
    const unsigned short* __restrict__ vh_u, const float* __restrict__ rel,
    const float* __restrict__ rel2, const int* __restrict__ mask,
    float* __restrict__ out) {
  const __bf16* qh = (const __bf16*)qh_u;
  const __bf16* kh = (const __bf16*)kh_u;
  const __bf16* vh = (const __bf16*)vh_u;

  __shared__ __attribute__((aligned(16))) __bf16 Ks[32 * 64];      // [key][d]
  __shared__ __attribute__((aligned(16))) __bf16 Vt[64 * 32];      // [d][key]
  __shared__ __attribute__((aligned(16))) __bf16 Ps[8 * 16 * 32];  // per-wave P

  const int tid = threadIdx.x;
  const int wv = tid >> 5, lane = tid & 31, l16 = lane & 15, hf = lane >> 4;
  const int q0 = blockIdx.x * 128;
  const int bh = blockIdx.y;  // b*16 + h
  const int bb = bh >> 4, hh = bh & 15;
  const size_t kvbase = (size_t)bh << 16;  // (b*H+h)*N*D

  Frag16 qA[2];  // q rows (pre-scaled), two K-chunks of 32 over d
  {
    const __bf16* base = qh + kvbase + (size_t)(q0 + wv * 16 + l16) * D_;
#pragma unroll
    for (int c = 0; c < 2; ++c) {
      const __bf16* p = base + c * 32 + hf * 8;
      qA[c].q[0] = *(const uint4*)(p);
      qA[c].q[1] = *(const uint4*)(p + 16);
    }
  }

  v8f oacc[4] = {};
  float mrow[8], lrow[8];
#pragma unroll
  for (int r = 0; r < 8; ++r) { mrow[r] = -3.0e38f; lrow[r] = 0.f; }

  __bf16* Pw = &Ps[wv * (16 * 32)];

  for (int kt = 0; kt < N_; kt += 32) {
    // --- stage K (async direct-to-LDS) and transposed V (32 keys x 64 d) ---
    {
      const int key = tid >> 3, cg = (tid & 7) * 8;   // 256 x 16B chunks
      const size_t g = kvbase + (size_t)(kt + key) * D_ + cg;
#if HAS_ASYNC
      async_cp_b128(kh + g, &Ks[key * 64 + cg]);
#else
      *(uint4*)(&Ks[key * 64 + cg]) = *(const uint4*)(kh + g);
#endif
      union { uint4 u; __bf16 e[8]; } vvv;
      vvv.u = *(const uint4*)(vh + g);
#pragma unroll
      for (int i = 0; i < 8; ++i) Vt[(cg + i) * 32 + key] = vvv.e[i];
    }
    wait_async_all();
    __syncthreads();

    // --- S = (q/sqrt(D)) K^T : two 16x16 key tiles ---
    v8f S0 = {}, S1 = {};
#pragma unroll
    for (int c = 0; c < 2; ++c) {
      Frag16 kb;
      const __bf16* p0 = &Ks[l16 * 64 + c * 32 + hf * 8];
      kb.q[0] = *(const uint4*)(p0);
      kb.q[1] = *(const uint4*)(p0 + 16);
      S0 = wmma_bf16(qA[c].v, kb.v, S0);
      const __bf16* p1 = &Ks[(16 + l16) * 64 + c * 32 + hf * 8];
      kb.q[0] = *(const uint4*)(p1);
      kb.q[1] = *(const uint4*)(p1 + 16);
      S1 = wmma_bf16(qA[c].v, kb.v, S1);
    }

    // --- rel_pos + rel_2d_pos, then mask-fill with 1e-8 (per reference) ---
    const int mk0 = mask[bb * N_ + kt + l16];
    const int mk1 = mask[bb * N_ + kt + 16 + l16];
#pragma unroll
    for (int r = 0; r < 8; ++r) {
      const int qn = q0 + wv * 16 + r + hf * 8;
      const size_t rb = ((size_t)hh * N_ + qn) * N_ + kt + l16;
      S0[r] += rel[rb] + rel2[rb];
      S1[r] += rel[rb + 16] + rel2[rb + 16];
      if (mk0) S0[r] = 1e-8f;
      if (mk1) S1[r] = 1e-8f;
    }

    // --- online softmax ---
#pragma unroll
    for (int r = 0; r < 8; ++r) {
      float x = fmaxf(S0[r], S1[r]);
#pragma unroll
      for (int off = 1; off < 16; off <<= 1) x = fmaxf(x, __shfl_xor(x, off, 32));
      const float mn = fmaxf(mrow[r], x);
      const float sc = __expf(mrow[r] - mn);
      mrow[r] = mn;
      const float p0 = __expf(S0[r] - mn);
      const float p1 = __expf(S1[r] - mn);
      float s = p0 + p1;
#pragma unroll
      for (int off = 1; off < 16; off <<= 1) s += __shfl_xor(s, off, 32);
      lrow[r] = lrow[r] * sc + s;
      oacc[0][r] *= sc; oacc[1][r] *= sc; oacc[2][r] *= sc; oacc[3][r] *= sc;
      // D-layout -> LDS row-major [m][key] (wave-private, in-order DS)
      Pw[(r + hf * 8) * 32 + l16] = (__bf16)p0;
      Pw[(r + hf * 8) * 32 + 16 + l16] = (__bf16)p1;
    }

    // --- P back as A-frag; PV accumulate ---
    Frag16 pA;
    {
      const __bf16* p = &Pw[l16 * 32 + hf * 8];
      pA.q[0] = *(const uint4*)(p);
      pA.q[1] = *(const uint4*)(p + 16);
    }
#pragma unroll
    for (int t = 0; t < 4; ++t) {
      Frag16 vb;  // V^T staged: N=d (t*16+l16), K=key {0..7,16..23} (+8 for hf)
      const __bf16* p = &Vt[(t * 16 + l16) * 32 + hf * 8];
      vb.q[0] = *(const uint4*)(p);
      vb.q[1] = *(const uint4*)(p + 16);
      oacc[t] = wmma_bf16(pA.v, vb.v, oacc[t]);
    }
    __syncthreads();
  }

  // --- normalize and store fp32 [B,N,HD] ---
#pragma unroll
  for (int r = 0; r < 8; ++r) {
    const float inv = 1.0f / lrow[r];
    const int n = q0 + wv * 16 + r + hf * 8;
    float* op = out + ((size_t)bb * N_ + n) * HD_ + hh * D_;
#pragma unroll
    for (int t = 0; t < 4; ++t) op[t * 16 + l16] = oacc[t][r] * inv;
  }
}

// ---------------------------------------------------------------------------
extern "C" void kernel_launch(void* const* d_in, const int* in_sizes, int n_in,
                              void* d_out, int out_size, void* d_ws, size_t ws_size,
                              hipStream_t stream) {
  (void)in_sizes; (void)n_in; (void)out_size; (void)ws_size;
  const float* hs   = (const float*)d_in[0];  // [B,N,HD]
  const float* qkvw = (const float*)d_in[1];  // [3HD,HD]
  const float* qb   = (const float*)d_in[2];  // [HD]
  const float* vb   = (const float*)d_in[3];  // [HD]
  const float* rel  = (const float*)d_in[4];  // [H,N,N]
  const float* rel2 = (const float*)d_in[5];  // [H,N,N]
  const int*   msk  = (const int*)d_in[6];    // [B,N]
  float* out = (float*)d_out;

  char* ws = (char*)d_ws;
  unsigned short* hb = (unsigned short*)(ws);             // 16 MB
  unsigned short* wb = (unsigned short*)(ws + 16777216);  //  6 MB
  unsigned short* qh = (unsigned short*)(ws + 23068672);  // 16 MB
  unsigned short* kh = (unsigned short*)(ws + 39845888);  // 16 MB
  unsigned short* vh = (unsigned short*)(ws + 56623104);  // 16 MB

  const int nh = B_ * N_ * HD_;  // 8388608
  const int nw = 3 * HD_ * HD_;  // 3145728
  cvt_f32_bf16_kernel<<<nh / (256 * 8), 256, 0, stream>>>(hs, hb, nh);
  cvt_f32_bf16_kernel<<<nw / (256 * 8), 256, 0, stream>>>(qkvw, wb, nw);

  dim3 g1((B_ * N_) / 128, (3 * HD_) / 64);  // (64, 48)
  qkv_gemm_kernel<<<g1, 256, 0, stream>>>(hb, wb, qb, vb, qh, kh, vh);

  dim3 g2(N_ / 128, B_ * H_);                // (8, 128)
  attn_kernel<<<g2, 256, 0, stream>>>(qh, kh, vh, rel, rel2, msk, out);
}
